// PairwiseRelationScorer_24257975287815
// MI455X (gfx1250) — matrix operand order repaired and checked by reading, not compiled, not run on hardware
//
#include <hip/hip_runtime.h>

#define B_    8
#define N_    256
#define H_    256
#define REL_  128
#define LANG_ 256

typedef __attribute__((ext_vector_type(16))) __bf16 v16bf;
typedef __attribute__((ext_vector_type(8)))  float  v8f;
typedef __attribute__((ext_vector_type(4)))  int    v4i_t;

#if __has_builtin(__builtin_amdgcn_global_load_async_to_lds_b128)
#define HAVE_ASYNC_COPY 1
typedef __attribute__((address_space(1))) v4i_t* gas_ptr_t;   // global source
typedef __attribute__((address_space(3))) v4i_t* lds_ptr_t;   // LDS destination
#define ASYNC_CP128(gsrc, ldst) \
    __builtin_amdgcn_global_load_async_to_lds_b128((gas_ptr_t)(gsrc), (lds_ptr_t)(ldst), 0, 0)
#else
#define HAVE_ASYNC_COPY 0
#endif

__device__ __forceinline__ void wait_async_then_barrier() {
#if HAVE_ASYNC_COPY
#if __has_builtin(__builtin_amdgcn_s_wait_asynccnt)
    __builtin_amdgcn_s_wait_asynccnt(0);
#else
    asm volatile("s_wait_asynccnt 0x0" ::: "memory");
#endif
#endif
    __syncthreads();
}

__device__ __forceinline__ unsigned short f32_to_bf16_bits(float f) {
    unsigned int u = __float_as_uint(f);
    unsigned int r = u + 0x7FFFu + ((u >> 16) & 1u);   // round-to-nearest-even
    return (unsigned short)(r >> 16);
}

// ---------------------------------------------------------------------------
// Kernel 0: convert object_h (B*N*H) and W1[0:2H, :] to bf16 bit-patterns.
// ---------------------------------------------------------------------------
__global__ void k_convert(const float* __restrict__ obj, const float* __restrict__ W1,
                          unsigned short* __restrict__ obj_bf,
                          unsigned short* __restrict__ w1_bf) {
    int t = blockIdx.x * blockDim.x + threadIdx.x;
    const int nobj = B_ * N_ * H_;       // 524288
    const int nw1  = 2 * H_ * H_;        // 131072
    if (t < nobj) {
        obj_bf[t] = f32_to_bf16_bits(obj[t]);
    } else {
        int u = t - nobj;
        if (u < nw1) w1_bf[u] = f32_to_bf16_bits(W1[u]);
    }
}

// ---------------------------------------------------------------------------
// Kernel 1: ar[b,h] = (rel_query[b] @ W_rel + b_rel) @ W1c + b1   (8 x 256)
// ---------------------------------------------------------------------------
__global__ void k_ar(const float* __restrict__ rel_query, const float* __restrict__ W_rel,
                     const float* __restrict__ b_rel, const float* __restrict__ W1,
                     const float* __restrict__ b1, float* __restrict__ ar) {
    __shared__ float rq[REL_];
    int b = blockIdx.x, tid = threadIdx.x;
    if (tid < REL_) {
        float s = b_rel[tid];
        #pragma unroll 4
        for (int l = 0; l < LANG_; ++l)
            s += rel_query[b * LANG_ + l] * W_rel[l * REL_ + tid];
        rq[tid] = s;
    }
    __syncthreads();
    float s = b1[tid];
    #pragma unroll 4
    for (int r = 0; r < REL_; ++r)
        s += rq[r] * W1[(2 * H_ + r) * H_ + tid];
    ar[b * H_ + tid] = s;
}

// ---------------------------------------------------------------------------
// Kernel 2: C[2048 x 256] = bf16(A) @ bf16(Bw) (+ bias row per b)
// Block = 8 waves. blockIdx encodes (row-group, column-strip).
// B^T strip (16 cols x 256 K) staged in LDS so each lane's B fragment is two
// contiguous ds_load_b128s. A fragment loaded straight from global (compiler
// merges the packed-pair dwords into global_load_b128).
// A and B fragments use contiguous k = 16*half + s, same order in both (wave32).
// ---------------------------------------------------------------------------
#define BT_STRIDE 272   // ushorts per B^T row: 256 + 16 pad (keeps 16B align, spreads banks)

__global__ void k_gemm_bf16(const unsigned short* __restrict__ A,
                            const unsigned short* __restrict__ Bw,
                            const float* __restrict__ bias,   // [B_ * H_] or nullptr
                            float* __restrict__ C) {
    __shared__ unsigned short sbt[16 * BT_STRIDE];   // B^T: [col 0..15][k 0..255]

    int tid = threadIdx.x;
    int tn  = blockIdx.x & 15;          // column strip (16 cols)
    int tmg = blockIdx.x >> 4;          // row group: 8 wave-tiles = 128 rows
    int colbase = tn * 16;

    // ---- stage B^T: thread t loads K-row t (16 contiguous bf16), scatters to LDS
    {
        const unsigned short* src = Bw + (size_t)tid * H_ + colbase;
        uint4 p0 = *(const uint4*)(src);
        uint4 p1 = *(const uint4*)(src + 8);
        unsigned short tmp[16];
        *(uint4*)(tmp)     = p0;
        *(uint4*)(tmp + 8) = p1;
        #pragma unroll
        for (int c = 0; c < 16; ++c)
            sbt[c * BT_STRIDE + tid] = tmp[c];
    }
    __syncthreads();

    int wave = tid >> 5;
    int lane = tid & 31;
    int half = lane >> 4;
    int mr   = lane & 15;
    int tm   = tmg * 8 + wave;          // 0..127
    int arow = tm * 16 + mr;            // A row for this lane
    int col  = colbase + mr;            // global output column

    v8f c = {0.f, 0.f, 0.f, 0.f, 0.f, 0.f, 0.f, 0.f};

    for (int k0 = 0; k0 < H_; k0 += 32) {
        // A fragment: packed K-pairs, dword loads (merged to b128 by compiler)
        union { v16bf v; unsigned int u[8]; } ua;
        #pragma unroll
        for (int v = 0; v < 8; ++v) {
            int kb = v * 2 + half * 16;
            ua.u[v] = *(const unsigned int*)(A + (size_t)arow * H_ + k0 + kb);
        }
        // B fragment: 16 contiguous K values for this lane's column -> 2x ds_load_b128
        union { v16bf v; uint4 q[2]; } ub;
        const unsigned short* bp = &sbt[mr * BT_STRIDE + k0 + half * 16];
        ub.q[0] = *(const uint4*)(bp);
        ub.q[1] = *(const uint4*)(bp + 8);

        // (neg_a, A, neg_b, B, c_mod, C, reuse_a, reuse_b)
        c = __builtin_amdgcn_wmma_f32_16x16x32_bf16(false, ua.v, false, ub.v,
                                                    (short)0, c, false, false);
    }

    float bv = bias ? bias[((tm * 16) >> 8) * H_ + col] : 0.f;  // b = row/256
    #pragma unroll
    for (int v = 0; v < 8; ++v) {
        int row = tm * 16 + v + 8 * half;
        C[(size_t)row * H_ + col] = c[v] + bv;
    }
}

// ---------------------------------------------------------------------------
// Kernel 3: out[b,i,j] = sum_h relu(ai[b,i,h] + ajr[b,j,h]) * W2[h] + b2
// 32(i) x 16(j) tile per block; tiles staged into LDS with async copies when
// the toolchain exposes GLOBAL_LOAD_ASYNC_TO_LDS_B128 (ASYNCcnt path).
// ---------------------------------------------------------------------------
#define TI 32
#define TJ 16
#define SSTRIDE 260   // 256 + 4 pad; 260*4 bytes keeps float4 alignment per row

__global__ void k_pair(const float* __restrict__ ai, const float* __restrict__ ajr,
                       const float* __restrict__ W2, const float* __restrict__ b2,
                       float* __restrict__ out) {
    __shared__ float sai[TI * SSTRIDE];
    __shared__ float sjr[TJ * SSTRIDE];
    __shared__ float sw2[H_];

    int b   = blockIdx.z;
    int i0g = blockIdx.x * TI;
    int j0g = blockIdx.y * TJ;
    int tid = threadIdx.x;

    const float4* aisrc = (const float4*)(ai + (size_t)(b * N_ + i0g) * H_);
    const float4* jrsrc = (const float4*)(ajr + (size_t)(b * N_ + j0g) * H_);

#if HAVE_ASYNC_COPY
    for (int idx = tid; idx < TI * (H_ / 4); idx += 256) {
        int r = idx >> 6, c4 = idx & 63;
        ASYNC_CP128(aisrc + r * (H_ / 4) + c4, &sai[r * SSTRIDE + c4 * 4]);
    }
    for (int idx = tid; idx < TJ * (H_ / 4); idx += 256) {
        int r = idx >> 6, c4 = idx & 63;
        ASYNC_CP128(jrsrc + r * (H_ / 4) + c4, &sjr[r * SSTRIDE + c4 * 4]);
    }
    if (tid < H_ / 4) {
        ASYNC_CP128(((const float4*)W2) + tid, &sw2[tid * 4]);
    }
#else
    for (int idx = tid; idx < TI * (H_ / 4); idx += 256) {
        int r = idx >> 6, c4 = idx & 63;
        *(float4*)(&sai[r * SSTRIDE + c4 * 4]) = aisrc[r * (H_ / 4) + c4];
    }
    for (int idx = tid; idx < TJ * (H_ / 4); idx += 256) {
        int r = idx >> 6, c4 = idx & 63;
        *(float4*)(&sjr[r * SSTRIDE + c4 * 4]) = jrsrc[r * (H_ / 4) + c4];
    }
    if (tid < H_ / 4) {
        *(float4*)(&sw2[tid * 4]) = ((const float4*)W2)[tid];
    }
#endif

    // prefetch the output rows we are about to write (global_prefetch_b8 path)
    __builtin_prefetch(out + (size_t)(b * N_ + i0g) * N_ + j0g, 1, 1);

    wait_async_then_barrier();

    int ti = tid >> 4;          // 0..15 -> i pair
    int tj = tid & 15;          // 0..15 -> j
    int i0 = ti * 2;

    const float* pa0 = &sai[i0 * SSTRIDE];
    const float* pa1 = &sai[(i0 + 1) * SSTRIDE];
    const float* pr  = &sjr[tj * SSTRIDE];

    float acc0 = 0.f, acc1 = 0.f;
    #pragma unroll 4
    for (int h = 0; h < H_; h += 4) {
        float4 a0 = *(const float4*)(pa0 + h);
        float4 a1 = *(const float4*)(pa1 + h);
        float4 r  = *(const float4*)(pr + h);
        float4 w  = *(const float4*)(&sw2[h]);
        acc0 = fmaf(fmaxf(a0.x + r.x, 0.f), w.x, acc0);
        acc0 = fmaf(fmaxf(a0.y + r.y, 0.f), w.y, acc0);
        acc0 = fmaf(fmaxf(a0.z + r.z, 0.f), w.z, acc0);
        acc0 = fmaf(fmaxf(a0.w + r.w, 0.f), w.w, acc0);
        acc1 = fmaf(fmaxf(a1.x + r.x, 0.f), w.x, acc1);
        acc1 = fmaf(fmaxf(a1.y + r.y, 0.f), w.y, acc1);
        acc1 = fmaf(fmaxf(a1.z + r.z, 0.f), w.z, acc1);
        acc1 = fmaf(fmaxf(a1.w + r.w, 0.f), w.w, acc1);
    }
    float bb = b2[0];
    out[(size_t)(b * N_ + i0g + i0)     * N_ + (j0g + tj)] = acc0 + bb;
    out[(size_t)(b * N_ + i0g + i0 + 1) * N_ + (j0g + tj)] = acc1 + bb;
}

// ---------------------------------------------------------------------------
// Launch
// ---------------------------------------------------------------------------
extern "C" void kernel_launch(void* const* d_in, const int* in_sizes, int n_in,
                              void* d_out, int out_size, void* d_ws, size_t ws_size,
                              hipStream_t stream) {
    const float* object_h  = (const float*)d_in[0];
    const float* rel_query = (const float*)d_in[1];
    const float* W_rel     = (const float*)d_in[2];
    const float* b_rel     = (const float*)d_in[3];
    const float* W1        = (const float*)d_in[4];
    const float* b1        = (const float*)d_in[5];
    const float* W2        = (const float*)d_in[6];
    const float* b2        = (const float*)d_in[7];
    float* out = (float*)d_out;

    // workspace layout (all offsets 256B-aligned), total ~5.3 MB
    char* ws = (char*)d_ws;
    unsigned short* obj_bf = (unsigned short*)(ws);            // 1,048,576 B
    unsigned short* w1_bf  = (unsigned short*)(ws + 1048576);  //   262,144 B
    float* ar  = (float*)(ws + 1310720);                       //     8,192 B
    float* ai  = (float*)(ws + 1318912);                       // 2,097,152 B
    float* ajr = (float*)(ws + 3416064);                       // 2,097,152 B

    // 0: bf16 conversion of activations + first 2H rows of W1
    k_convert<<<(B_*N_*H_ + 2*H_*H_ + 255) / 256, 256, 0, stream>>>(object_h, W1, obj_bf, w1_bf);

    // 1: ar = (rel_query @ W_rel + b_rel) @ W1c + b1
    k_ar<<<B_, 256, 0, stream>>>(rel_query, W_rel, b_rel, W1, b1, ar);

    // 2: ai = object_h @ W1a  (WMMA bf16, f32 accum)
    k_gemm_bf16<<<256, 256, 0, stream>>>(obj_bf, w1_bf, nullptr, ai);
    // 3: ajr = object_h @ W1b + ar  (bias fused)
    k_gemm_bf16<<<256, 256, 0, stream>>>(obj_bf, w1_bf + (size_t)H_ * H_, ar, ajr);

    // 4: fused pairwise relu-reduce
    dim3 grid(N_ / TI, N_ / TJ, B_);
    k_pair<<<grid, 256, 0, stream>>>(ai, ajr, W2, b2, out);
}
